// EDTransformer_21887153341005
// MI455X (gfx1250) — matrix operand order repaired and checked by reading, not compiled, not run on hardware
//
#include <hip/hip_runtime.h>
#include <hip/hip_bf16.h>

// ---------------------------------------------------------------------------
// Model constants (from reference)
// ---------------------------------------------------------------------------
#define BSZ   4
#define LSEQ  512          // LX == LZ == 512
#define DE    1024
#define DMLP  4096
#define NH    16
#define DHEAD 64
#define NVOC  32000
#define EPSV  1e-5f

// Use the CDNA5 Tensor Data Mover for GEMM tile staging (6-arg clang-23 builtin).
#define USE_TDM 1

typedef __attribute__((ext_vector_type(16))) __bf16 v16bf;
typedef __attribute__((ext_vector_type(8)))  float  v8f;
typedef __attribute__((ext_vector_type(4)))  unsigned int v4u;
typedef __attribute__((ext_vector_type(8)))  int v8i;
typedef __attribute__((ext_vector_type(4)))  int v4i;

union FragU { unsigned int u[8]; v16bf v; };

// Pack two fp32 -> two bf16 (round-to-nearest-ish) in one dword:
// low half = bf16(lo), high half = bf16(hi).  2 adds + 1 v_perm_b32.
__device__ __forceinline__ unsigned packbf(float lo, float hi) {
    unsigned a = __float_as_uint(hi) + 0x8000u;
    unsigned b = __float_as_uint(lo) + 0x8000u;
    return __builtin_amdgcn_perm(a, b, 0x07060302u);
}

#if USE_TDM
// Issue one 2D TDM tile load: tile1 rows x tile0 contiguous elements (fp32),
// row pitch stride0 elements, into LDS at byte offset ldsOff.
__device__ __forceinline__ void tdm_load_2d(const float* gptr, unsigned ldsOff,
                                            unsigned tile0, unsigned tile1,
                                            unsigned long long stride0)
{
    unsigned long long ga = (unsigned long long)gptr;
    unsigned td0 = (unsigned)stride0;      // tensor_dim0: full row length
    unsigned td1 = 1u << 22;               // tensor_dim1: large (no OOB clipping)
    v4u g0;
    g0[0] = 1u;                                              // count=1, user mode
    g0[1] = ldsOff;                                          // lds_addr (bytes)
    g0[2] = (unsigned)ga;                                    // global_addr[31:0]
    g0[3] = (unsigned)((ga >> 32) & 0x01FFFFFFu) | 0x80000000u; // addr[56:32] | type=2
    v8i g1;
    g1[0] = 0x00020000;                                      // data_size = 2 (4 bytes)
    g1[1] = (int)((td0 & 0xFFFFu) << 16);                    // tensor_dim0[15:0]
    g1[2] = (int)(((td0 >> 16) & 0xFFFFu) | ((td1 & 0xFFFFu) << 16));
    g1[3] = (int)(((td1 >> 16) & 0xFFFFu) | ((tile0 & 0xFFFFu) << 16)); // tile_dim0
    g1[4] = (int)(tile1 & 0xFFFFu);                          // tile_dim1 (tile_dim2=0)
    g1[5] = (int)(unsigned)(stride0 & 0xFFFFFFFFull);        // tensor_dim0_stride lo
    g1[6] = (int)(unsigned)((stride0 >> 32) & 0xFFFFull);    // stride hi
    g1[7] = 0;
    v4i gz4; gz4[0] = 0; gz4[1] = 0; gz4[2] = 0; gz4[3] = 0;
    v8i gz8; for (int i = 0; i < 8; ++i) gz8[i] = 0;
    __builtin_amdgcn_tensor_load_to_lds(g0, g1, gz4, gz4, gz8, 0);
}
#endif

// ---------------------------------------------------------------------------
// Generic mixed-precision GEMM:  C[m,n] = sum_k A[m,k]*B[k,n] (+bias[m]) (+res) (relu?)
// Tile 64x64, BK=32, 128 threads (4 waves), each wave does a 32x32 quadrant
// with 2x2 V_WMMA_F32_16X16X32_BF16 accumulators.  fp32 tiles are DMA'd into
// LDS by the Tensor Data Mover, repacked to bf16 pairs with v_perm, then fed
// to the WMMAs via the documented CDNA5 fragment layouts.
// flags: bit0 = A stored transposed (A_st[k*lda+m]), bit1 = ReLU epilogue.
// Requires: gridDim.x*64 == M, gridDim.y*64 == N, K % 32 == 0.
// ---------------------------------------------------------------------------
#define LAP 17   // padded row pitch (dwords) of A bf16 tile: conflict-free col reads

__global__ __launch_bounds__(128) void gemm64x64_bf16wmma(
    const float* __restrict__ A, const float* __restrict__ B, float* __restrict__ C,
    const float* __restrict__ bias, const float* __restrict__ res,
    int K, int lda, int ldb, int ldc,
    long long sA, long long sB, long long sC, long long sRes, int flags)
{
    const int bz = blockIdx.z;
    A += (long long)bz * sA;
    B += (long long)bz * sB;
    C += (long long)bz * sC;
    if (res) res += (long long)bz * sRes;

    const int m0   = blockIdx.x * 64;
    const int n0   = blockIdx.y * 64;
    const int t    = threadIdx.x;
    const int wave = t >> 5;
    const int lane = t & 31;
    const int half = lane >> 4;
    const int ln   = lane & 15;
    const int wm   = (wave >> 1) * 32;   // wave row offset inside tile
    const int wn   = (wave & 1) * 32;    // wave col offset inside tile
    const bool transA = (flags & 1) != 0;

    __shared__ alignas(16) float    stA[64 * 32];      // fp32 staging (TDM dest)
    __shared__ alignas(16) float    stB[32 * 64];
    __shared__ alignas(16) unsigned lAu[64 * LAP];     // bf16 pairs [m][kpair]
    __shared__ alignas(16) unsigned lBu[16 * 64];      // bf16 pairs [kpair][n]

    v8f acc[2][2];
    for (int i = 0; i < 2; ++i)
        for (int j = 0; j < 2; ++j)
            for (int r = 0; r < 8; ++r) acc[i][j][r] = 0.0f;

#if USE_TDM
    const unsigned offA = (unsigned)(unsigned long long)(void*)stA;  // LDS byte offset
    const unsigned offB = (unsigned)(unsigned long long)(void*)stB;
#endif

    for (int k0 = 0; k0 < K; k0 += 32) {
#if USE_TDM
        if (t < 32) {   // wave 0 drives the DMA
            const float* aSrc = transA ? (A + (long long)k0 * lda + m0)
                                       : (A + (long long)m0 * lda + k0);
            unsigned at0 = transA ? 64u : 32u;   // contiguous extent
            unsigned at1 = transA ? 32u : 64u;   // rows
            tdm_load_2d(aSrc, offA, at0, at1, (unsigned long long)lda);
            tdm_load_2d(B + (long long)k0 * ldb + n0, offB, 64u, 32u,
                        (unsigned long long)ldb);
            __builtin_amdgcn_s_wait_tensorcnt(0);
        }
        __syncthreads();   // fp32 staging tiles visible to all waves
        // ---- repack staging fp32 -> bf16 pair tiles ----
        if (!transA) {     // stA is [m][k] row-major; pairs contiguous
            const float2* a2 = (const float2*)stA;
            for (int r = 0; r < 8; ++r) {
                int e = t + r * 128;                 // pair index: m*16+p
                float2 f = a2[e];
                lAu[(e >> 4) * LAP + (e & 15)] = packbf(f.x, f.y);
            }
        } else {           // stA is [k][m] (32x64)
            for (int r = 0; r < 8; ++r) {
                int e = t + r * 128; int m = e & 63, p = e >> 6;
                lAu[m * LAP + p] = packbf(stA[(2 * p) * 64 + m], stA[(2 * p + 1) * 64 + m]);
            }
        }
        for (int r = 0; r < 8; ++r) {
            int e = t + r * 128; int q = e >> 6, n = e & 63;
            lBu[q * 64 + n] = packbf(stB[(2 * q) * 64 + n], stB[(2 * q + 1) * 64 + n]);
        }
#else
        // Fallback: direct global loads + perm pack.
        if (!transA) {
            for (int r = 0; r < 8; ++r) {
                int e = t + r * 128; int m = e >> 4, p = e & 15;
                const float2 f = *(const float2*)(A + (long long)(m0 + m) * lda + k0 + 2 * p);
                lAu[m * LAP + p] = packbf(f.x, f.y);
            }
        } else {
            for (int r = 0; r < 8; ++r) {
                int e = t + r * 128; int m = e & 63, p = e >> 6;
                float lo = A[(long long)(k0 + 2 * p) * lda + (m0 + m)];
                float hi = A[(long long)(k0 + 2 * p + 1) * lda + (m0 + m)];
                lAu[m * LAP + p] = packbf(lo, hi);
            }
        }
        for (int r = 0; r < 8; ++r) {
            int e = t + r * 128; int q = e >> 6, n = e & 63;
            float lo = B[(long long)(k0 + 2 * q) * ldb + (n0 + n)];
            float hi = B[(long long)(k0 + 2 * q + 1) * ldb + (n0 + n)];
            lBu[q * 64 + n] = packbf(lo, hi);
        }
        if (k0 + 32 < K)
            __builtin_prefetch(&B[(long long)(k0 + 32) * ldb + n0 + (t & 63)], 0, 1);
#endif
        __syncthreads();   // bf16 tiles ready

        // ---- fragments per documented CDNA5 VGPR layouts ----
        FragU af[2], bf[2];
        for (int i = 0; i < 2; ++i) {
            int mrow = wm + i * 16 + ln;
            for (int j = 0; j < 8; ++j) {
                int pair = (j < 4) ? (half * 4 + j) : (8 + half * 4 + (j - 4));
                af[i].u[j] = lAu[mrow * LAP + pair];
            }
        }
        for (int j = 0; j < 2; ++j) {
            int ncol = wn + j * 16 + ln;
            for (int v = 0; v < 8; ++v)
                bf[j].u[v] = lBu[((v < 4) ? (half * 4 + v) : (8 + half * 4 + (v - 4))) * 64 + ncol];
        }
        for (int i = 0; i < 2; ++i)
            for (int j = 0; j < 2; ++j)
                acc[i][j] = __builtin_amdgcn_wmma_f32_16x16x32_bf16(
                    false, af[i].v, false, bf[j].v, (short)0, acc[i][j], false, false);
        __syncthreads();   // fragment reads done before next overwrite
    }

    // ---- epilogue: bias / residual / relu, fp32 store ----
    const bool doRelu = (flags & 2) != 0;
    for (int i = 0; i < 2; ++i) {
        for (int j = 0; j < 2; ++j) {
            int n = n0 + wn + j * 16 + ln;
            for (int r = 0; r < 8; ++r) {
                int m = m0 + wm + i * 16 + half * 8 + r;
                float v = acc[i][j][r];
                if (bias) v += bias[m];
                if (res)  v += res[(long long)m * ldc + n];
                if (doRelu) v = fmaxf(v, 0.0f);
                C[(long long)m * ldc + n] = v;
            }
        }
    }
}

// ---------------------------------------------------------------------------
// Embedding: O[b,d,l] = We[tok[b,l],d] + Wp[l,d];  one block per (b,l)
// ---------------------------------------------------------------------------
__global__ __launch_bounds__(256) void embed_kernel(
    float* __restrict__ O, const float* __restrict__ We, const float* __restrict__ Wp,
    const int* __restrict__ tok, int L)
{
    int bl = blockIdx.x, bb = bl / L, l = bl % L;
    int tk = tok[bl];
    for (int d = threadIdx.x; d < DE; d += 256)
        O[((long long)bb * DE + d) * L + l] = We[(long long)tk * DE + d] + Wp[(long long)l * DE + d];
}

// ---------------------------------------------------------------------------
// LayerNorm over feature dim (unbiased var, ddof=1); one block per (b,l)
// ---------------------------------------------------------------------------
__global__ __launch_bounds__(256) void ln_kernel(
    const float* __restrict__ E, float* __restrict__ O,
    const float* __restrict__ g, const float* __restrict__ b, int L)
{
    __shared__ float rs[256], rq[256];
    int bl = blockIdx.x, bb = bl / L, l = bl % L;
    const float* e = E + (long long)bb * DE * L + l;
    float* o = O + (long long)bb * DE * L + l;
    int t = threadIdx.x;
    float s = 0.0f, ss = 0.0f;
    for (int d = t; d < DE; d += 256) {
        float x = e[(long long)d * L];
        s += x; ss += x * x;
    }
    rs[t] = s; rq[t] = ss; __syncthreads();
    for (int off = 128; off > 0; off >>= 1) {
        if (t < off) { rs[t] += rs[t + off]; rq[t] += rq[t + off]; }
        __syncthreads();
    }
    float sum = rs[0], sq = rq[0];
    float mean = sum / (float)DE;
    float var  = (sq - sum * sum / (float)DE) / (float)(DE - 1);
    float inv  = rsqrtf(var + EPSV);
    for (int d = t; d < DE; d += 256) {
        float x = e[(long long)d * L];
        o[(long long)d * L] = (x - mean) * inv * g[d] + b[d];
    }
}

// ---------------------------------------------------------------------------
// Softmax over keys (dim z) of S[bh, z, q], scale 1/sqrt(64); optional causal
// mask (key z valid iff z <= q). One block per (q, bh). In place.
// ---------------------------------------------------------------------------
__global__ __launch_bounds__(256) void softmax_keys_kernel(
    float* __restrict__ S, int Lz, int Lq, int causal)
{
    __shared__ float red[256];
    int q = blockIdx.x, t = threadIdx.x;
    float* col = S + (long long)blockIdx.y * Lz * Lq + q;
    int lim = causal ? (q + 1) : Lz;
    float mx = -3.4e38f;
    for (int z = t; z < lim; z += 256)
        mx = fmaxf(mx, col[(long long)z * Lq] * 0.125f);
    red[t] = mx; __syncthreads();
    for (int off = 128; off > 0; off >>= 1) {
        if (t < off) red[t] = fmaxf(red[t], red[t + off]);
        __syncthreads();
    }
    mx = red[0]; __syncthreads();
    float sum = 0.0f;
    for (int z = t; z < lim; z += 256)
        sum += __expf(col[(long long)z * Lq] * 0.125f - mx);
    red[t] = sum; __syncthreads();
    for (int off = 128; off > 0; off >>= 1) {
        if (t < off) red[t] += red[t + off];
        __syncthreads();
    }
    float inv = 1.0f / red[0];
    for (int z = t; z < Lz; z += 256) {
        float o = (z < lim) ? __expf(col[(long long)z * Lq] * 0.125f - mx) * inv : 0.0f;
        col[(long long)z * Lq] = o;
    }
}

// ---------------------------------------------------------------------------
// Softmax over vocab dim of Out[b, v, l]; one block per (b,l). In place.
// ---------------------------------------------------------------------------
__global__ __launch_bounds__(256) void softmax_vocab_kernel(float* __restrict__ Out, int L)
{
    __shared__ float red[256];
    int bl = blockIdx.x, bb = bl / L, l = bl % L, t = threadIdx.x;
    float* col = Out + (long long)bb * NVOC * L + l;
    float mx = -3.4e38f;
    for (int v = t; v < NVOC; v += 256) mx = fmaxf(mx, col[(long long)v * L]);
    red[t] = mx; __syncthreads();
    for (int off = 128; off > 0; off >>= 1) {
        if (t < off) red[t] = fmaxf(red[t], red[t + off]);
        __syncthreads();
    }
    mx = red[0]; __syncthreads();
    float sum = 0.0f;
    for (int v = t; v < NVOC; v += 256) sum += __expf(col[(long long)v * L] - mx);
    red[t] = sum; __syncthreads();
    for (int off = 128; off > 0; off >>= 1) {
        if (t < off) red[t] += red[t + off];
        __syncthreads();
    }
    float inv = 1.0f / red[0];
    for (int v = t; v < NVOC; v += 256)
        col[(long long)v * L] = __expf(col[(long long)v * L] - mx) * inv;
}

// ---------------------------------------------------------------------------
// Host orchestration
// ---------------------------------------------------------------------------
static void launch_gemm(hipStream_t st, const float* A, const float* B, float* C,
                        const float* bias, const float* res,
                        int M, int N, int K, int lda, int ldb, int ldc,
                        long long sA, long long sB, long long sC, long long sRes,
                        int batches, int flags)
{
    dim3 grid(M / 64, N / 64, batches);
    gemm64x64_bf16wmma<<<grid, 128, 0, st>>>(A, B, C, bias, res, K, lda, ldb, ldc,
                                             sA, sB, sC, sRes, flags);
}

extern "C" void kernel_launch(void* const* d_in, const int* in_sizes, int n_in,
                              void* d_out, int out_size, void* d_ws, size_t ws_size,
                              hipStream_t stream)
{
    (void)in_sizes; (void)n_in; (void)out_size; (void)ws_size;

    const int*   xt = (const int*)d_in[0];
    const int*   zt = (const int*)d_in[1];
    const float* We = (const float*)d_in[2];
    const float* Wp = (const float*)d_in[3];
    const float* Wu = (const float*)d_in[4];
    const float* eWq = (const float*)d_in[5],  *ebq = (const float*)d_in[6];
    const float* eWk = (const float*)d_in[7],  *ebk = (const float*)d_in[8];
    const float* eWv = (const float*)d_in[9],  *ebv = (const float*)d_in[10];
    const float* eWo = (const float*)d_in[11], *ebo = (const float*)d_in[12];
    const float* eW1 = (const float*)d_in[13], *eb1 = (const float*)d_in[14];
    const float* eW2 = (const float*)d_in[15], *eb2 = (const float*)d_in[16];
    const float* eg1 = (const float*)d_in[17], *eB1 = (const float*)d_in[18];
    const float* eg2 = (const float*)d_in[19], *eB2 = (const float*)d_in[20];
    const float* dWq = (const float*)d_in[21], *dbq = (const float*)d_in[22];
    const float* dWk = (const float*)d_in[23], *dbk = (const float*)d_in[24];
    const float* dWv = (const float*)d_in[25], *dbv = (const float*)d_in[26];
    const float* dWo = (const float*)d_in[27], *dbo = (const float*)d_in[28];
    const float* dW1 = (const float*)d_in[29], *db1 = (const float*)d_in[30];
    const float* dW2 = (const float*)d_in[31], *db2 = (const float*)d_in[32];
    const float* dg1 = (const float*)d_in[33], *dB1b = (const float*)d_in[34];
    const float* dg2 = (const float*)d_in[35], *dB2b = (const float*)d_in[36];
    const float* dg3 = (const float*)d_in[37], *dB3b = (const float*)d_in[38];

    // workspace layout (floats)
    const long long ACT = (long long)BSZ * DE * LSEQ;       // 2,097,152
    float* w  = (float*)d_ws;
    float* Z  = w;
    float* X  = Z + ACT;
    float* Q  = X + ACT;
    float* Kb = Q + ACT;
    float* V  = Kb + ACT;
    float* Y  = V + ACT;
    float* S  = Y + ACT;                                     // 4*16*512*512
    float* T1 = S + (long long)BSZ * NH * LSEQ * LSEQ;       // 4*4096*512
    float* Out = (float*)d_out;

    const long long DEL = (long long)DE * LSEQ;    // per-batch activation stride
    const long long QBH = (long long)DHEAD * LSEQ; // per-(b,h) q/k/v/Y stride
    const long long SBH = (long long)LSEQ * LSEQ;  // per-(b,h) score stride
    const long long WQKV_L = (long long)NH * DHEAD * DE;
    const long long WO_L = (long long)DE * NH * DHEAD;
    const long long W1_L = (long long)DMLP * DE, W2_L = (long long)DE * DMLP;

    dim3 lnGrid(BSZ * LSEQ);
    dim3 smGrid(LSEQ, BSZ * NH);

    // ---- embeddings ----
    embed_kernel<<<dim3(BSZ * LSEQ), 256, 0, stream>>>(Z, We, Wp, zt, LSEQ);
    embed_kernel<<<dim3(BSZ * LSEQ), 256, 0, stream>>>(X, We, Wp, xt, LSEQ);

    // ---- encoder ----
    for (int l = 0; l < 2; ++l) {
        const float* Wq = eWq + l * WQKV_L; const float* bq = ebq + l * (NH * DHEAD);
        const float* Wk = eWk + l * WQKV_L; const float* bk = ebk + l * (NH * DHEAD);
        const float* Wv = eWv + l * WQKV_L; const float* bv = ebv + l * (NH * DHEAD);
        const float* Wo = eWo + l * WO_L;   const float* bo = ebo + l * DE;
        launch_gemm(stream, Wq, Z, Q,  bq, nullptr, NH*DHEAD, LSEQ, DE, DE, LSEQ, LSEQ, 0, DEL, DEL, 0, BSZ, 0);
        launch_gemm(stream, Wk, Z, Kb, bk, nullptr, NH*DHEAD, LSEQ, DE, DE, LSEQ, LSEQ, 0, DEL, DEL, 0, BSZ, 0);
        launch_gemm(stream, Wv, Z, V,  bv, nullptr, NH*DHEAD, LSEQ, DE, DE, LSEQ, LSEQ, 0, DEL, DEL, 0, BSZ, 0);
        launch_gemm(stream, Kb, Q, S, nullptr, nullptr, LSEQ, LSEQ, DHEAD, LSEQ, LSEQ, LSEQ,
                    QBH, QBH, SBH, 0, BSZ * NH, 1);
        softmax_keys_kernel<<<smGrid, 256, 0, stream>>>(S, LSEQ, LSEQ, 0);
        launch_gemm(stream, V, S, Y, nullptr, nullptr, DHEAD, LSEQ, LSEQ, LSEQ, LSEQ, LSEQ,
                    QBH, SBH, QBH, 0, BSZ * NH, 0);
        launch_gemm(stream, Wo, Y, T1, bo, Z, DE, LSEQ, NH*DHEAD, NH*DHEAD, LSEQ, LSEQ,
                    0, DEL, DEL, DEL, BSZ, 0);
        ln_kernel<<<lnGrid, 256, 0, stream>>>(T1, Z, eg1 + l * DE, eB1 + l * DE, LSEQ);
        launch_gemm(stream, eW1 + l * W1_L, Z, T1, eb1 + l * DMLP, nullptr,
                    DMLP, LSEQ, DE, DE, LSEQ, LSEQ, 0, DEL, (long long)DMLP * LSEQ, 0, BSZ, 2);
        launch_gemm(stream, eW2 + l * W2_L, T1, Q, eb2 + l * DE, Z,
                    DE, LSEQ, DMLP, DMLP, LSEQ, LSEQ, 0, (long long)DMLP * LSEQ, DEL, DEL, BSZ, 0);
        ln_kernel<<<lnGrid, 256, 0, stream>>>(Q, Z, eg2 + l * DE, eB2 + l * DE, LSEQ);
    }

    // ---- decoder ----
    for (int l = 0; l < 2; ++l) {
        const float* Wq = dWq + l * WQKV_L; const float* bq = dbq + l * (NH * DHEAD);
        const float* Wk = dWk + l * WQKV_L; const float* bk = dbk + l * (NH * DHEAD);
        const float* Wv = dWv + l * WQKV_L; const float* bv = dbv + l * (NH * DHEAD);
        const float* Wo = dWo + l * WO_L;   const float* bo = dbo + l * DE;
        // self-attention (causal)
        launch_gemm(stream, Wq, X, Q,  bq, nullptr, NH*DHEAD, LSEQ, DE, DE, LSEQ, LSEQ, 0, DEL, DEL, 0, BSZ, 0);
        launch_gemm(stream, Wk, X, Kb, bk, nullptr, NH*DHEAD, LSEQ, DE, DE, LSEQ, LSEQ, 0, DEL, DEL, 0, BSZ, 0);
        launch_gemm(stream, Wv, X, V,  bv, nullptr, NH*DHEAD, LSEQ, DE, DE, LSEQ, LSEQ, 0, DEL, DEL, 0, BSZ, 0);
        launch_gemm(stream, Kb, Q, S, nullptr, nullptr, LSEQ, LSEQ, DHEAD, LSEQ, LSEQ, LSEQ,
                    QBH, QBH, SBH, 0, BSZ * NH, 1);
        softmax_keys_kernel<<<smGrid, 256, 0, stream>>>(S, LSEQ, LSEQ, 1);
        launch_gemm(stream, V, S, Y, nullptr, nullptr, DHEAD, LSEQ, LSEQ, LSEQ, LSEQ, LSEQ,
                    QBH, SBH, QBH, 0, BSZ * NH, 0);
        launch_gemm(stream, Wo, Y, T1, bo, X, DE, LSEQ, NH*DHEAD, NH*DHEAD, LSEQ, LSEQ,
                    0, DEL, DEL, DEL, BSZ, 0);
        ln_kernel<<<lnGrid, 256, 0, stream>>>(T1, X, dg1 + l * DE, dB1b + l * DE, LSEQ);
        // cross-attention (same weights, K/V from encoder output Z)
        launch_gemm(stream, Wq, X, Q,  bq, nullptr, NH*DHEAD, LSEQ, DE, DE, LSEQ, LSEQ, 0, DEL, DEL, 0, BSZ, 0);
        launch_gemm(stream, Wk, Z, Kb, bk, nullptr, NH*DHEAD, LSEQ, DE, DE, LSEQ, LSEQ, 0, DEL, DEL, 0, BSZ, 0);
        launch_gemm(stream, Wv, Z, V,  bv, nullptr, NH*DHEAD, LSEQ, DE, DE, LSEQ, LSEQ, 0, DEL, DEL, 0, BSZ, 0);
        launch_gemm(stream, Kb, Q, S, nullptr, nullptr, LSEQ, LSEQ, DHEAD, LSEQ, LSEQ, LSEQ,
                    QBH, QBH, SBH, 0, BSZ * NH, 1);
        softmax_keys_kernel<<<smGrid, 256, 0, stream>>>(S, LSEQ, LSEQ, 0);
        launch_gemm(stream, V, S, Y, nullptr, nullptr, DHEAD, LSEQ, LSEQ, LSEQ, LSEQ, LSEQ,
                    QBH, SBH, QBH, 0, BSZ * NH, 0);
        launch_gemm(stream, Wo, Y, T1, bo, X, DE, LSEQ, NH*DHEAD, NH*DHEAD, LSEQ, LSEQ,
                    0, DEL, DEL, DEL, BSZ, 0);
        ln_kernel<<<lnGrid, 256, 0, stream>>>(T1, X, dg2 + l * DE, dB2b + l * DE, LSEQ);
        // MLP
        launch_gemm(stream, dW1 + l * W1_L, X, T1, db1 + l * DMLP, nullptr,
                    DMLP, LSEQ, DE, DE, LSEQ, LSEQ, 0, DEL, (long long)DMLP * LSEQ, 0, BSZ, 2);
        launch_gemm(stream, dW2 + l * W2_L, T1, Q, db2 + l * DE, X,
                    DE, LSEQ, DMLP, DMLP, LSEQ, LSEQ, 0, (long long)DMLP * LSEQ, DEL, DEL, BSZ, 0);
        ln_kernel<<<lnGrid, 256, 0, stream>>>(Q, X, dg3 + l * DE, dB3b + l * DE, LSEQ);
    }

    // ---- unembedding + vocab softmax (logits written straight to d_out) ----
    launch_gemm(stream, Wu, X, Out, nullptr, nullptr, NVOC, LSEQ, DE, DE, LSEQ, LSEQ,
                0, DEL, (long long)NVOC * LSEQ, 0, BSZ, 0);
    softmax_vocab_kernel<<<dim3(BSZ * LSEQ), 256, 0, stream>>>(Out, LSEQ);
}
